// RAGA_73839077752944
// MI455X (gfx1250) — matrix operand order repaired and checked by reading, not compiled, not run on hardware
//
#include <hip/hip_runtime.h>
#include <hip/hip_bf16.h>
#include <math.h>

#define EH   300
#define RH   100
#define KPAD 320            // K padded to 10 * 32
#define COLT 19             // ceil(300/16) output col tiles
#define WPAD (COLT * 16)    // 304 padded weight columns
#define OUTC 2000

typedef __attribute__((ext_vector_type(16))) __bf16 v16bf;
typedef __attribute__((ext_vector_type(8)))  float  v8f;

__device__ __forceinline__ float lrelu01(float v) { return v > 0.f ? v : 0.01f * v; }

// monotone float <-> uint mapping for atomicMax-based segment max
__device__ __forceinline__ unsigned f2ord(float f) {
    unsigned b = __float_as_uint(f);
    return (b & 0x80000000u) ? ~b : (b | 0x80000000u);
}
__device__ __forceinline__ float ord2f(unsigned u) {
    return (u & 0x80000000u) ? __uint_as_float(u ^ 0x80000000u) : __uint_as_float(~u);
}

// ---------------- degree / norm ----------------
__global__ void k_deg(const int* __restrict__ dst, float* deg, int E) {
    int e = blockIdx.x * blockDim.x + threadIdx.x;
    if (e < E) atomicAdd(&deg[dst[e]], 1.0f);
}
__global__ void k_dis(const float* __restrict__ deg, float* dis, int n) {
    int t = blockIdx.x * blockDim.x + threadIdx.x;
    if (t < n) dis[t] = deg[t] > 0.f ? rsqrtf(deg[t]) : 0.f;
}

// ---------------- GCN neighborhood aggregate (scatter-add) ----------------
__global__ void k_gcn_agg(const float* __restrict__ x, const int* __restrict__ ej,
                          const int* __restrict__ ei, const float* __restrict__ dis,
                          float* agg, int E) {
    int e = blockIdx.x, d = threadIdx.x;
    if (e >= E || d >= EH) return;
    int j = ej[e], i = ei[e];
    float nrm = dis[j] * dis[i];
    atomicAdd(&agg[(size_t)i * EH + d], nrm * x[(size_t)j * EH + d]);
}

// ---------------- bf16 staging ----------------
__global__ void k_to_bf16_pad(const float* __restrict__ in, __bf16* out,
                              int rows, int cols, int ldo, int do_relu) {
    size_t t = (size_t)blockIdx.x * blockDim.x + threadIdx.x;
    size_t total = (size_t)rows * ldo;
    if (t >= total) return;
    int r = (int)(t / ldo), k = (int)(t % ldo);
    float v = 0.f;
    if (k < cols) { v = in[(size_t)r * cols + k]; if (do_relu) v = fmaxf(v, 0.f); }
    out[t] = (__bf16)v;
}
// wt[n*ldk + k] = w[k*Ncol + n]   (transpose so B loads are K-contiguous)
__global__ void k_w_to_bf16_t(const float* __restrict__ w, __bf16* wt,
                              int K, int Ncol, int ldk, int rowsOut) {
    size_t t = (size_t)blockIdx.x * blockDim.x + threadIdx.x;
    size_t total = (size_t)rowsOut * ldk;
    if (t >= total) return;
    int n = (int)(t / ldk), k = (int)(t % ldk);
    float v = (k < K && n < Ncol) ? w[(size_t)k * Ncol + n] : 0.f;
    wt[t] = (__bf16)v;
}

// ---------------- WMMA GEMM: C[rows,outCols] = A[rows,ldk] * Bt^T ----------------
__global__ void k_wmma_gemm(const __bf16* __restrict__ A, const __bf16* __restrict__ Bt,
                            float* __restrict__ C, int rowTiles, int colTiles,
                            int ldk, int outCols) {
    int wave = blockIdx.x * (blockDim.x >> 5) + (threadIdx.x >> 5);
    int lane = threadIdx.x & 31;
    int tiles = rowTiles * colTiles;
    if (wave >= tiles) return;                 // wave-uniform: EXEC stays all-1s
    int m0 = (wave / colTiles) * 16;
    int n0 = (wave % colTiles) * 16;
    int half = lane >> 4, mr = lane & 15;
    const __bf16* Arow = A  + (size_t)(m0 + mr) * ldk;
    const __bf16* Bcol = Bt + (size_t)(n0 + mr) * ldk;
    v8f acc = {};
    for (int k0 = 0; k0 < ldk; k0 += 32) {
        v16bf a, b;
        const __bf16* p0 = Arow + k0 + half * 8;       // K = k0 + half*8 + 0..7
        const __bf16* p1 = Arow + k0 + 16 + half * 8;  // K = k0+16 + half*8 + 0..7
        #pragma unroll
        for (int q = 0; q < 8; q++) { a[q] = p0[q]; a[8 + q] = p1[q]; }
        const __bf16* pb0 = Bcol + k0 + half * 8;      // K = k0 + half*8 + 0..7
        const __bf16* pb1 = Bcol + k0 + 16 + half * 8; // K = k0+16 + half*8 + 0..7
        #pragma unroll
        for (int q = 0; q < 8; q++) { b[q] = pb0[q]; b[8 + q] = pb1[q]; }
        acc = __builtin_amdgcn_wmma_f32_16x16x32_bf16(false, a, false, b,
                                                      (short)0, acc, false, false);
    }
    int col = n0 + mr;
    if (col < outCols) {
        #pragma unroll
        for (int r = 0; r < 8; r++)
            C[(size_t)(m0 + r + half * 8) * outCols + col] = acc[r];
    }
}

// ---------------- highway combine ----------------
__global__ void k_highway(const float* __restrict__ x1, const float* __restrict__ x2,
                          const float* __restrict__ gl, const float* __restrict__ b,
                          float* xout, int n) {
    size_t t = (size_t)blockIdx.x * blockDim.x + threadIdx.x;
    if (t >= (size_t)n * EH) return;
    int d = (int)(t % EH);
    float g = 1.f / (1.f + expf(-(gl[t] + b[d])));
    xout[t] = g * x2[t] + (1.f - g) * x1[t];
}

// ---------------- wave-per-row matvec ----------------
__global__ void k_matvec(const float* __restrict__ X, const float* __restrict__ v,
                         float* out, int rows, int cols, int ldx) {
    int wave = blockIdx.x * (blockDim.x >> 5) + (threadIdx.x >> 5);
    int lane = threadIdx.x & 31;
    if (wave >= rows) return;
    const float* xr = X + (size_t)wave * ldx;
    float s = 0.f;
    for (int c = lane; c < cols; c += 32) s += xr[c] * v[c];
    for (int off = 16; off; off >>= 1) s += __shfl_down(s, off, 32);
    if (lane == 0) out[wave] = s;
}

// ---------------- elementwise ----------------
__global__ void k_lrelu(const float* __restrict__ in, float* out, size_t n) {
    size_t t = (size_t)blockIdx.x * blockDim.x + threadIdx.x;
    if (t < n) out[t] = lrelu01(in[t]);
}
__global__ void k_relu_copy(const float* __restrict__ in, float* out, size_t n) {
    size_t t = (size_t)blockIdx.x * blockDim.x + threadIdx.x;
    if (t < n) out[t] = fmaxf(in[t], 0.f);
}
__global__ void k_copy_x(const float* __restrict__ x, float* out, int n) {
    size_t t = (size_t)blockIdx.x * blockDim.x + threadIdx.x;
    if (t >= (size_t)n * EH) return;
    out[(t / EH) * OUTC + (t % EH)] = x[t];
}
__global__ void k_relu_gat(float* out, int n) {
    size_t t = (size_t)blockIdx.x * blockDim.x + threadIdx.x;
    if (t >= (size_t)n * 1000) return;
    size_t idx = (t / 1000) * OUTC + 1000 + (t % 1000);
    out[idx] = fmaxf(out[idx], 0.f);
}

// ---------------- _gat_r: score lrelu(si[i]+sj[j]), softmax over j, agg into i ----------------
__global__ void k_relr_max(const int* ej, const int* ei, const float* si, const float* sj,
                           unsigned* mu, int E) {
    int e = blockIdx.x * blockDim.x + threadIdx.x; if (e >= E) return;
    int j = ej[e], i = ei[e];
    atomicMax(&mu[j], f2ord(lrelu01(si[i] + sj[j])));
}
__global__ void k_relr_sum(const int* ej, const int* ei, const float* si, const float* sj,
                           const unsigned* mu, float* ssum, int E) {
    int e = blockIdx.x * blockDim.x + threadIdx.x; if (e >= E) return;
    int j = ej[e], i = ei[e];
    float sc = lrelu01(si[i] + sj[j]);
    atomicAdd(&ssum[j], expf(sc - ord2f(mu[j])));
}
__global__ void k_relr_agg(const int* ej, const int* ei, const float* si, const float* sj,
                           const unsigned* mu, const float* ssum,
                           const float* __restrict__ emb, float* agg, int E) {
    int e = blockIdx.x, d = threadIdx.x;
    if (e >= E || d >= RH) return;
    int j = ej[e], i = ei[e];
    float sc = lrelu01(si[i] + sj[j]);
    float alpha = expf(sc - ord2f(mu[j])) / (ssum[j] + 1e-16f);
    atomicAdd(&agg[(size_t)i * RH + d], alpha * emb[(size_t)j * RH + d]);
}

// ---------------- wjq: score dI[i]+dR[rel]+dJ[j] (no lrelu), softmax & agg over i ----------------
__global__ void k_wjq_max(const int* ei0, const int* ei1, const int* rel,
                          const float* dI, const float* dJ, const float* dR,
                          unsigned* mu, int E) {
    int e = blockIdx.x * blockDim.x + threadIdx.x; if (e >= E) return;
    int i = ei0[e], j = ei1[e];
    atomicMax(&mu[i], f2ord(dI[i] + dR[rel[e]] + dJ[j]));
}
__global__ void k_wjq_sum(const int* ei0, const int* ei1, const int* rel,
                          const float* dI, const float* dJ, const float* dR,
                          const unsigned* mu, float* ssum, int E) {
    int e = blockIdx.x * blockDim.x + threadIdx.x; if (e >= E) return;
    int i = ei0[e], j = ei1[e];
    float sc = dI[i] + dR[rel[e]] + dJ[j];
    atomicAdd(&ssum[i], expf(sc - ord2f(mu[i])));
}
__global__ void k_wjq_agg(const int* ei0, const int* ei1, const int* rel,
                          const float* dI, const float* dJ, const float* dR,
                          const unsigned* mu, const float* ssum,
                          const float* __restrict__ ef, const float* __restrict__ re,
                          float* out, int E) {
    int e = blockIdx.x, d = threadIdx.x;
    if (e >= E || d >= 700) return;
    int i = ei0[e], j = ei1[e], r = rel[e];
    float sc = dI[i] + dR[r] + dJ[j];
    float att = expf(sc - ord2f(mu[i])) / (ssum[i] + 1e-16f);
    float fv = (d < 300) ? ef[(size_t)i * EH + d]
             : (d < 400) ? re[(size_t)r * RH + (d - 300)]
                         : ef[(size_t)j * EH + (d - 400)];
    atomicAdd(&out[(size_t)i * OUTC + 300 + d], att * fv);
}

// ---------------- final GAT: score lrelu(sI[i]+sJ[j]+sr[rel]), softmax & agg over i ----------------
__global__ void k_gat_max(const int* ej0, const int* ei1, const int* rel,
                          const float* sI, const float* sJ, const float* sr,
                          unsigned* mu, int E) {
    int e = blockIdx.x * blockDim.x + threadIdx.x; if (e >= E) return;
    int j = ej0[e], i = ei1[e];
    atomicMax(&mu[i], f2ord(lrelu01(sI[i] + sJ[j] + sr[rel[e]])));
}
__global__ void k_gat_sum(const int* ej0, const int* ei1, const int* rel,
                          const float* sI, const float* sJ, const float* sr,
                          const unsigned* mu, float* ssum, int E) {
    int e = blockIdx.x * blockDim.x + threadIdx.x; if (e >= E) return;
    int j = ej0[e], i = ei1[e];
    float sc = lrelu01(sI[i] + sJ[j] + sr[rel[e]]);
    atomicAdd(&ssum[i], expf(sc - ord2f(mu[i])));
}
__global__ void k_gat_agg(const int* ej0, const int* ei1, const int* rel,
                          const float* sI, const float* sJ, const float* sr,
                          const unsigned* mu, const float* ssum, float* out, int E) {
    int e = blockIdx.x, d = threadIdx.x;
    if (e >= E || d >= 1000) return;
    int j = ej0[e], i = ei1[e];
    float sc = lrelu01(sI[i] + sJ[j] + sr[rel[e]]);
    float alpha = expf(sc - ord2f(mu[i])) / (ssum[i] + 1e-16f);
    atomicAdd(&out[(size_t)i * OUTC + 1000 + d], alpha * out[(size_t)j * OUTC + d]);
}

extern "C" void kernel_launch(void* const* d_in, const int* in_sizes, int n_in,
                              void* d_out, int out_size, void* d_ws, size_t ws_size,
                              hipStream_t stream) {
    const float* x_e      = (const float*)d_in[0];
    const int*   eia      = (const int*)d_in[3];
    const int*   rel_all  = (const int*)d_in[4];
    const int*   lgo      = (const int*)d_in[5];
    const int*   lgi      = (const int*)d_in[7];
    const float* gcn1_w   = (const float*)d_in[9];
    const float* gcn2_w   = (const float*)d_in[10];
    const float* hw1_w    = (const float*)d_in[11];
    const float* hw1_b    = (const float*)d_in[12];
    const float* hw2_w    = (const float*)d_in[13];
    const float* hw2_b    = (const float*)d_in[14];
    const float* ww1      = (const float*)d_in[15];
    const float* gat_ai   = (const float*)d_in[16];
    const float* gat_aj   = (const float*)d_in[17];
    const float* gat_ar   = (const float*)d_in[18];
    const float* gatr_ai  = (const float*)d_in[19];
    const float* gatr_aj  = (const float*)d_in[20];
    const float* rel_emb1 = (const float*)d_in[21];

    const int N   = in_sizes[0] / EH;
    const int E   = in_sizes[4];
    const int ELG = in_sizes[6];
    const int R   = in_sizes[21] / RH;
    const int R2  = 2 * R;
    float* out = (float*)d_out;

    const int* eia_j = eia;       const int* eia_i = eia + E;
    const int* lgo_j = lgo;       const int* lgo_i = lgo + ELG;
    const int* lgi_j = lgi;       const int* lgi_i = lgi + ELG;

    // workspace carve-out
    size_t off = 0;
    auto carve = [&](size_t bytes) -> void* {
        void* p = (char*)d_ws + off;
        off += (bytes + 255) & ~(size_t)255;
        return p;
    };
    float*    deg  = (float*)carve((size_t)N * 4);
    float*    dis  = (float*)carve((size_t)N * 4);
    float*    agg  = (float*)carve((size_t)N * EH * 4);
    float*    go1  = (float*)carve((size_t)N * EH * 4);   // gcn gemm out
    float*    go2  = (float*)carve((size_t)N * EH * 4);   // gate gemm out
    float*    xcur = (float*)carve((size_t)N * EH * 4);
    float*    ef   = (float*)carve((size_t)N * EH * 4);
    __bf16*   Abf  = (__bf16*)carve((size_t)N * KPAD * 2);
    __bf16*   Wbf  = (__bf16*)carve((size_t)WPAD * KPAD * 2);
    float*    si   = (float*)carve((size_t)R * 4);
    float*    sj   = (float*)carve((size_t)R * 4);
    float*    ragg = (float*)carve((size_t)R * RH * 4);
    float*    remb = (float*)carve((size_t)R2 * RH * 4);  // concat rel_emb [2R,100]
    float*    re   = (float*)carve((size_t)R2 * RH * 4);
    float*    dR2  = (float*)carve((size_t)R2 * 4);
    float*    srv  = (float*)carve((size_t)R2 * 4);
    float*    dI   = (float*)carve((size_t)N * 4);
    float*    dJ   = (float*)carve((size_t)N * 4);
    float*    sIn  = (float*)carve((size_t)N * 4);
    float*    sJn  = (float*)carve((size_t)N * 4);
    unsigned* mu   = (unsigned*)carve((size_t)N * 4);
    float*    ssum = (float*)carve((size_t)N * 4);

    const int rowTiles = (N + 15) / 16;
    const int tiles = rowTiles * COLT;
    dim3 gemmGrid((tiles + 3) / 4), gemmBlk(128);
    size_t nPad = (size_t)N * KPAD, wPad = (size_t)WPAD * KPAD;
    int gPadA = (int)((nPad + 255) / 256), gPadW = (int)((wPad + 255) / 256);
    int gNE = (int)(((size_t)N * EH + 255) / 256);
    int gE256 = (E + 255) / 256, gELG256 = (ELG + 255) / 256;

    // ---- degree / norm (shared by both GCN layers) ----
    hipMemsetAsync(deg, 0, (size_t)N * 4, stream);
    k_deg<<<gE256, 256, 0, stream>>>(eia_i, deg, E);
    k_dis<<<(N + 255) / 256, 256, 0, stream>>>(deg, dis, N);
    hipMemsetAsync(out, 0, (size_t)N * OUTC * 4, stream);

    // ---- layer 1: x = highway(x_e, gcn(x_e)) ----
    hipMemsetAsync(agg, 0, (size_t)N * EH * 4, stream);
    k_gcn_agg<<<E, 320, 0, stream>>>(x_e, eia_j, eia_i, dis, agg, E);
    k_w_to_bf16_t<<<gPadW, 256, 0, stream>>>(gcn1_w, Wbf, EH, EH, KPAD, WPAD);
    k_to_bf16_pad<<<gPadA, 256, 0, stream>>>(agg, Abf, N, EH, KPAD, 1);
    k_wmma_gemm<<<gemmGrid, gemmBlk, 0, stream>>>(Abf, Wbf, go1, rowTiles, COLT, KPAD, EH);
    k_w_to_bf16_t<<<gPadW, 256, 0, stream>>>(hw1_w, Wbf, EH, EH, KPAD, WPAD);
    k_to_bf16_pad<<<gPadA, 256, 0, stream>>>(x_e, Abf, N, EH, KPAD, 0);
    k_wmma_gemm<<<gemmGrid, gemmBlk, 0, stream>>>(Abf, Wbf, go2, rowTiles, COLT, KPAD, EH);
    k_highway<<<gNE, 256, 0, stream>>>(x_e, go1, go2, hw1_b, xcur, N);

    // ---- layer 2: x = highway(x, gcn(x)) ----
    hipMemsetAsync(agg, 0, (size_t)N * EH * 4, stream);
    k_gcn_agg<<<E, 320, 0, stream>>>(xcur, eia_j, eia_i, dis, agg, E);
    k_w_to_bf16_t<<<gPadW, 256, 0, stream>>>(gcn2_w, Wbf, EH, EH, KPAD, WPAD);
    k_to_bf16_pad<<<gPadA, 256, 0, stream>>>(agg, Abf, N, EH, KPAD, 1);
    k_wmma_gemm<<<gemmGrid, gemmBlk, 0, stream>>>(Abf, Wbf, go1, rowTiles, COLT, KPAD, EH);
    k_w_to_bf16_t<<<gPadW, 256, 0, stream>>>(hw2_w, Wbf, EH, EH, KPAD, WPAD);
    k_to_bf16_pad<<<gPadA, 256, 0, stream>>>(xcur, Abf, N, EH, KPAD, 0);
    k_wmma_gemm<<<gemmGrid, gemmBlk, 0, stream>>>(Abf, Wbf, go2, rowTiles, COLT, KPAD, EH);
    k_highway<<<gNE, 256, 0, stream>>>(xcur, go1, go2, hw2_b, xcur, N);  // in-place (elementwise)

    // ---- _gat_r on the two line graphs -> remb [2R, RH] ----
    k_matvec<<<(R + 3) / 4, 128, 0, stream>>>(rel_emb1, gatr_ai, si, R, RH, RH);
    k_matvec<<<(R + 3) / 4, 128, 0, stream>>>(rel_emb1, gatr_aj, sj, R, RH, RH);
    const int* lg_j[2] = {lgo_j, lgi_j};
    const int* lg_i[2] = {lgo_i, lgi_i};
    for (int g = 0; g < 2; g++) {
        hipMemsetAsync(mu,   0, (size_t)R * 4, stream);
        hipMemsetAsync(ssum, 0, (size_t)R * 4, stream);
        hipMemsetAsync(ragg, 0, (size_t)R * RH * 4, stream);
        k_relr_max<<<gELG256, 256, 0, stream>>>(lg_j[g], lg_i[g], si, sj, mu, ELG);
        k_relr_sum<<<gELG256, 256, 0, stream>>>(lg_j[g], lg_i[g], si, sj, mu, ssum, ELG);
        k_relr_agg<<<ELG, 128, 0, stream>>>(lg_j[g], lg_i[g], si, sj, mu, ssum, rel_emb1, ragg, ELG);
        k_relu_copy<<<(int)(((size_t)R * RH + 255) / 256), 256, 0, stream>>>(
            ragg, remb + (size_t)g * R * RH, (size_t)R * RH);
    }

    // ---- _graphattention_wjq -> out cols [300,1000) ; out cols [0,300) = x ----
    k_lrelu<<<gNE, 256, 0, stream>>>(xcur, ef, (size_t)N * EH);
    k_lrelu<<<(int)(((size_t)R2 * RH + 255) / 256), 256, 0, stream>>>(remb, re, (size_t)R2 * RH);
    k_matvec<<<(N + 3) / 4, 128, 0, stream>>>(ef, ww1,       dI,  N,  EH, EH);
    k_matvec<<<(N + 3) / 4, 128, 0, stream>>>(ef, ww1 + 400, dJ,  N,  EH, EH);
    k_matvec<<<(R2 + 3) / 4, 128, 0, stream>>>(re, ww1 + 300, dR2, R2, RH, RH);
    hipMemsetAsync(mu,   0, (size_t)N * 4, stream);
    hipMemsetAsync(ssum, 0, (size_t)N * 4, stream);
    k_wjq_max<<<gE256, 256, 0, stream>>>(eia_j, eia_i, rel_all, dI, dJ, dR2, mu, E);
    k_wjq_sum<<<gE256, 256, 0, stream>>>(eia_j, eia_i, rel_all, dI, dJ, dR2, mu, ssum, E);
    k_copy_x<<<gNE, 256, 0, stream>>>(xcur, out, N);
    k_wjq_agg<<<E, 704, 0, stream>>>(eia_j, eia_i, rel_all, dI, dJ, dR2, mu, ssum, ef, re, out, E);

    // ---- final _gat over x_wjq (= out cols [0,1000)) -> out cols [1000,2000) ----
    k_matvec<<<(N + 3) / 4, 128, 0, stream>>>(out, gat_ai, sIn, N, 1000, OUTC);
    k_matvec<<<(N + 3) / 4, 128, 0, stream>>>(out, gat_aj, sJn, N, 1000, OUTC);
    k_matvec<<<(R2 + 3) / 4, 128, 0, stream>>>(remb, gat_ar, srv, R2, RH, RH);
    hipMemsetAsync(mu,   0, (size_t)N * 4, stream);
    hipMemsetAsync(ssum, 0, (size_t)N * 4, stream);
    k_gat_max<<<gE256, 256, 0, stream>>>(eia_j, eia_i, rel_all, sIn, sJn, srv, mu, E);
    k_gat_sum<<<gE256, 256, 0, stream>>>(eia_j, eia_i, rel_all, sIn, sJn, srv, mu, ssum, E);
    k_gat_agg<<<E, 1024, 0, stream>>>(eia_j, eia_i, rel_all, sIn, sJn, srv, mu, ssum, out, E);
    k_relu_gat<<<(int)(((size_t)N * 1000 + 255) / 256), 256, 0, stream>>>(out, N);
    (void)ws_size; (void)n_in; (void)out_size;
}
